// PoseEncoderLayer_12292196401619
// MI455X (gfx1250) — compile-verified, matrix operand order checked by reading, not run on hardware
//
#include <hip/hip_runtime.h>
#include <hip/hip_bf16.h>

// ---------------------------------------------------------------------------
// PoseEncoderLayer on gfx1250: bf16 WMMA GEMMs with fused epilogues.
// B=8 N=512 DIM=1024 HEADS=16 HD=64 MLP=4096 E=8 PART=256
// WG tile 128x64, 8 waves, each wave 32x32 (4x v_wmma_f32_16x16x32_bf16 / Kstep)
// ---------------------------------------------------------------------------

typedef __attribute__((ext_vector_type(16))) __bf16 v16bf;
typedef __attribute__((ext_vector_type(8)))  __bf16 v8bf;
typedef __attribute__((ext_vector_type(8)))  float  v8f;

#define TB   8
#define TN   512
#define TDIM 1024
#define THEADS 16
#define THD  64
#define TMLP 4096
#define TE   8
#define TPART 256

__device__ __forceinline__ __bf16 f2bf(float f) {
    unsigned u = __builtin_bit_cast(unsigned, f);
    unsigned r = (u + 0x7FFFu + ((u >> 16) & 1u)) >> 16;
    return __builtin_bit_cast(__bf16, (unsigned short)r);
}

// Build a v16bf WMMA fragment from an LDS row laid out K-contiguous.
// VGPR r holds K = (r<4 ? 2r : 2r+8) + half*8 and +1  =>  lo = [half*8, +8),
// hi = [16+half*8, +8): two aligned 16-byte LDS loads.
__device__ __forceinline__ v16bf load_frag(const __bf16* rowp, int half) {
    v8bf lo = *(const v8bf*)(rowp + half * 8);
    v8bf hi = *(const v8bf*)(rowp + 16 + half * 8);
    return __builtin_shufflevector(lo, hi, 0, 1, 2, 3, 4, 5, 6, 7,
                                   8, 9, 10, 11, 12, 13, 14, 15);
}

// ---------------------------------------------------------------------------
// f32 -> bf16 cast (weights)
// ---------------------------------------------------------------------------
__global__ void cast_bf16_kernel(const float* __restrict__ in,
                                 __bf16* __restrict__ out, long n) {
    long i = (long)blockIdx.x * blockDim.x + threadIdx.x;
    if (i < n) out[i] = f2bf(in[i]);
}

// ---------------------------------------------------------------------------
// LayerNorm over last dim (dim=1024), one block per row, bf16 output
// ---------------------------------------------------------------------------
__global__ void layernorm_bf16_kernel(const float* __restrict__ x,
                                      const float* __restrict__ w,
                                      const float* __restrict__ b,
                                      __bf16* __restrict__ out, int dim) {
    const int tid = threadIdx.x;
    const long row = blockIdx.x;
    const float* xr = x + row * (long)dim;
    __shared__ float red[256];

    float s = 0.f;
    for (int i = tid; i < dim; i += 256) s += xr[i];
    red[tid] = s; __syncthreads();
    for (int o = 128; o > 0; o >>= 1) { if (tid < o) red[tid] += red[tid + o]; __syncthreads(); }
    const float mean = red[0] / dim;
    __syncthreads();

    float v = 0.f;
    for (int i = tid; i < dim; i += 256) { float d = xr[i] - mean; v += d * d; }
    red[tid] = v; __syncthreads();
    for (int o = 128; o > 0; o >>= 1) { if (tid < o) red[tid] += red[tid + o]; __syncthreads(); }
    const float rstd = rsqrtf(red[0] / dim + 1e-5f);

    __bf16* orow = out + row * (long)dim;
    for (int i = tid; i < dim; i += 256)
        orow[i] = f2bf((xr[i] - mean) * rstd * w[i] + b[i]);
}

// ---------------------------------------------------------------------------
// Row softmax over 512 elements, in place (f32). One block per row.
// ---------------------------------------------------------------------------
__global__ void softmax512_kernel(float* __restrict__ p) {
    const int tid = threadIdx.x;
    float* row = p + (long)blockIdx.x * 512;
    float a0 = row[tid], a1 = row[tid + 256];
    __shared__ float red[256];

    red[tid] = fmaxf(a0, a1); __syncthreads();
    for (int o = 128; o > 0; o >>= 1) { if (tid < o) red[tid] = fmaxf(red[tid], red[tid + o]); __syncthreads(); }
    const float mx = red[0];
    __syncthreads();

    const float e0 = expf(a0 - mx), e1 = expf(a1 - mx);
    red[tid] = e0 + e1; __syncthreads();
    for (int o = 128; o > 0; o >>= 1) { if (tid < o) red[tid] += red[tid + o]; __syncthreads(); }
    const float inv = 1.f / red[0];
    row[tid] = e0 * inv;
    row[tid + 256] = e1 * inv;
}

// ---------------------------------------------------------------------------
// Generic batched bf16 WMMA GEMM:  C = epilogue(alpha * A @ B)
//   grid = (N/64, M/128, batch); 256 threads = 8 waves
//   LDS: A tile 128x32 (row-major), B tile stored TRANSPOSED as Bs[64 n][32 k]
//   wave (wm,wn) owns rows [wm*32,+32) x cols [wn*32,+32): 4 WMMAs / K-step
// ---------------------------------------------------------------------------
template<int TRANS_B, int A_F32, int HAS_BIAS, int GELU, int RES, int OUT_BF16>
__global__ __launch_bounds__(256)
void gemm_wmma_kernel(const void* __restrict__ A, int lda, long asB, long asH,
                      const __bf16* __restrict__ Bm, int ldb, long bsB, long bsH,
                      const int* __restrict__ bIdx,
                      void* __restrict__ C, int ldc, long csB, long csH,
                      const float* __restrict__ bias, long biasStride,
                      const float* __restrict__ Rsd, int ldr, long rsB,
                      int M, int N, int K, int nh, float alpha) {
    const int tid  = threadIdx.x;
    const int z    = blockIdx.z;
    const int zb   = z / nh, zh = z % nh;

    const long aoff = (long)zb * asB + (long)zh * asH;
    const long boff = bIdx ? (long)bIdx[z] * bsB : (long)zb * bsB + (long)zh * bsH;
    const long coff = (long)zb * csB + (long)zh * csH;

    const int wave = tid >> 5;
    const int wm   = wave & 3;   // 0..3 -> 32-row sub-tile
    const int wn   = wave >> 2;  // 0..1 -> 32-col sub-tile
    const int lane = tid & 31;
    const int half = lane >> 4;
    const int mr   = lane & 15;

    __shared__ __bf16 At[128][40];     // [m][k]
    __shared__ __bf16 Bs[64][40];      // [n][k]  (transposed tile)

    v8f acc[2][2] = {};
    const long bn0 = (long)blockIdx.x * 64;
    const long am0 = (long)blockIdx.y * 128;

    for (int k0 = 0; k0 < K; k0 += 32) {
        // ---- stage A tile 128x32: 16 contiguous K elems per thread ----
        {
            const int row = tid >> 1;
            const int kc  = (tid & 1) * 16;
            const long gr = am0 + row;
            if (A_F32) {
                const float* Ap = (const float*)A + aoff;
                #pragma unroll
                for (int j = 0; j < 16; ++j)
                    At[row][kc + j] = f2bf(Ap[gr * (long)lda + k0 + kc + j]);
            } else {
                const __bf16* Ap = (const __bf16*)A + aoff;
                #pragma unroll
                for (int j = 0; j < 16; ++j)
                    At[row][kc + j] = Ap[gr * (long)lda + k0 + kc + j];
            }
        }
        // ---- stage B tile (32 k x 64 n) transposed into Bs[n][k] ----
        if (TRANS_B) {
            // memory is [N, K] row-major: contiguous along k -> vector stores
            const int nn = tid >> 2;            // 0..63
            const int kc = (tid & 3) * 8;       // 0,8,16,24
            const __bf16* Bp = Bm + boff;
            #pragma unroll
            for (int j = 0; j < 8; ++j)
                Bs[nn][kc + j] = Bp[(bn0 + nn) * (long)ldb + k0 + kc + j];
        } else {
            // memory is [K, N] row-major: coalesced read along n, scatter to LDS
            const int kk  = tid >> 3;           // 0..31
            const int nn0 = (tid & 7) * 8;      // 0..56
            const __bf16* Bp = Bm + boff;
            #pragma unroll
            for (int j = 0; j < 8; ++j)
                Bs[nn0 + j][kk] = Bp[(long)(k0 + kk) * ldb + bn0 + nn0 + j];
        }
        __syncthreads();

        // ---- fragments + 4 WMMAs (2 A frags x 2 B frags) ----
        v16bf af[2], bf[2];
        #pragma unroll
        for (int s = 0; s < 2; ++s) {
            af[s] = load_frag(&At[wm * 32 + s * 16 + mr][0], half);
            bf[s] = load_frag(&Bs[wn * 32 + s * 16 + mr][0], half);
        }
        #pragma unroll
        for (int sm = 0; sm < 2; ++sm)
            #pragma unroll
            for (int sn = 0; sn < 2; ++sn)
                acc[sm][sn] = __builtin_amdgcn_wmma_f32_16x16x32_bf16(
                    false, af[sm], false, bf[sn], (short)0, acc[sm][sn],
                    false, false);
        __syncthreads();
    }

    // ---- epilogue ----
    const float* biasp = HAS_BIAS
        ? bias + (bIdx ? (long)bIdx[z] * biasStride : 0)
        : nullptr;
    const float* Rp = RES ? Rsd + (long)zb * rsB : nullptr;

    #pragma unroll
    for (int sm = 0; sm < 2; ++sm) {
        #pragma unroll
        for (int sn = 0; sn < 2; ++sn) {
            const int colg = (int)bn0 + wn * 32 + sn * 16 + mr;
            #pragma unroll
            for (int r = 0; r < 8; ++r) {
                const long rowg = am0 + wm * 32 + sm * 16 + r + half * 8;
                float v = acc[sm][sn][r] * alpha;
                if (HAS_BIAS) v += biasp[colg];
                if (GELU)     v = 0.5f * v * (1.f + erff(v * 0.70710678118f));
                if (RES)      v += Rp[rowg * (long)ldr + colg];
                if (OUT_BF16) ((__bf16*)C)[coff + rowg * (long)ldc + colg] = f2bf(v);
                else          ((float*) C)[coff + rowg * (long)ldc + colg] = v;
            }
        }
    }
}

// ---------------------------------------------------------------------------
// Host-side orchestration
// ---------------------------------------------------------------------------
extern "C" void kernel_launch(void* const* d_in, const int* in_sizes, int n_in,
                              void* d_out, int out_size, void* d_ws, size_t ws_size,
                              hipStream_t stream) {
    (void)in_sizes; (void)n_in; (void)out_size; (void)ws_size;

    const float* x      = (const float*)d_in[0];
    const int*   idx    = (const int*)  d_in[1];
    const float* ln1_w  = (const float*)d_in[2];
    const float* ln1_b  = (const float*)d_in[3];
    const float* qkv_w  = (const float*)d_in[4];
    const float* qkv_b  = (const float*)d_in[5];
    const float* out_w  = (const float*)d_in[6];
    const float* out_b  = (const float*)d_in[7];
    const float* ln2_w  = (const float*)d_in[8];
    const float* ln2_b  = (const float*)d_in[9];
    const float* fc1_w  = (const float*)d_in[10];
    const float* fc1_b  = (const float*)d_in[11];
    const float* fc2_w  = (const float*)d_in[12];
    const float* fc2_b  = (const float*)d_in[13];
    const float* exp_w  = (const float*)d_in[14];
    const float* exp_b  = (const float*)d_in[15];

    const long M   = (long)TB * TN;         // 4096
    const long D3  = 3L * TDIM;             // 3072
    float* outFinal = (float*)d_out;                         // [B,N,DIM]
    float* attn     = outFinal + M * TDIM;                   // [B,H,N,N]

    // workspace carve-up (256B aligned)
    size_t o = 0;
    auto carve = [&](size_t bytes) {
        size_t r = o; o = (o + bytes + 255) & ~(size_t)255; return r;
    };
    char* ws = (char*)d_ws;
    __bf16* qkvWb  = (__bf16*)(ws + carve((size_t)TDIM * D3 * 2));
    __bf16* outWb  = (__bf16*)(ws + carve((size_t)TDIM * TDIM * 2));
    __bf16* fc1Wb  = (__bf16*)(ws + carve((size_t)TDIM * TMLP * 2));
    __bf16* fc2Wb  = (__bf16*)(ws + carve((size_t)TMLP * (TDIM - TPART) * 2));
    __bf16* expWb  = (__bf16*)(ws + carve((size_t)TE * TMLP * TPART * 2));
    __bf16* xn1    = (__bf16*)(ws + carve((size_t)M * TDIM * 2));
    __bf16* qkv16  = (__bf16*)(ws + carve((size_t)M * D3 * 2));
    __bf16* aout16 = (__bf16*)(ws + carve((size_t)M * TDIM * 2));
    float*  x2     = (float*) (ws + carve((size_t)M * TDIM * 4));
    __bf16* xn2    = (__bf16*)(ws + carve((size_t)M * TDIM * 2));
    __bf16* h16    = (__bf16*)(ws + carve((size_t)M * TMLP * 2));

    auto cast = [&](const float* src, __bf16* dst, long n) {
        cast_bf16_kernel<<<dim3((unsigned)((n + 255) / 256)), 256, 0, stream>>>(src, dst, n);
    };
    cast(qkv_w, qkvWb, (long)TDIM * D3);
    cast(out_w, outWb, (long)TDIM * TDIM);
    cast(fc1_w, fc1Wb, (long)TDIM * TMLP);
    cast(fc2_w, fc2Wb, (long)TMLP * (TDIM - TPART));
    cast(exp_w, expWb, (long)TE * TMLP * TPART);

    // LN1
    layernorm_bf16_kernel<<<dim3((unsigned)M), 256, 0, stream>>>(x, ln1_w, ln1_b, xn1, TDIM);

    // QKV: [4096,1024] @ [1024,3072] + b -> bf16
    gemm_wmma_kernel<0,0,1,0,0,1><<<dim3(D3/64, M/128, 1), 256, 0, stream>>>(
        xn1, TDIM, 0, 0,  qkvWb, (int)D3, 0, 0, nullptr,
        qkv16, (int)D3, 0, 0,  qkv_b, 0,  nullptr, 0, 0,
        (int)M, (int)D3, TDIM, 1, 1.0f);

    // dots = Q @ K^T * DIM^-0.5  -> attn region (f32), batch B*H
    gemm_wmma_kernel<1,0,0,0,0,0><<<dim3(TN/64, TN/128, TB*THEADS), 256, 0, stream>>>(
        qkv16,          (int)D3, (long)TN*D3, THD,
        qkv16 + TDIM,   (int)D3, (long)TN*D3, THD, nullptr,
        attn, TN, (long)THEADS*TN*TN, (long)TN*TN,
        nullptr, 0,  nullptr, 0, 0,
        TN, TN, THD, THEADS, 0.03125f);

    // softmax rows (in place on d_out attn region)
    softmax512_kernel<<<dim3((unsigned)(TB*THEADS*TN)), 256, 0, stream>>>(attn);

    // attn @ V -> bf16 [B,N,DIM] head-interleaved
    gemm_wmma_kernel<0,1,0,0,0,1><<<dim3(THD/64, TN/128, TB*THEADS), 256, 0, stream>>>(
        attn, TN, (long)THEADS*TN*TN, (long)TN*TN,
        qkv16 + 2*TDIM, (int)D3, (long)TN*D3, THD, nullptr,
        aout16, TDIM, (long)TN*TDIM, THD,
        nullptr, 0,  nullptr, 0, 0,
        TN, THD, TN, THEADS, 1.0f);

    // out proj + bias + residual(x) -> x2 (f32)
    gemm_wmma_kernel<0,0,1,0,1,0><<<dim3(TDIM/64, M/128, 1), 256, 0, stream>>>(
        aout16, TDIM, 0, 0,  outWb, TDIM, 0, 0, nullptr,
        x2, TDIM, 0, 0,  out_b, 0,  x, TDIM, 0,
        (int)M, TDIM, TDIM, 1, 1.0f);

    // LN2
    layernorm_bf16_kernel<<<dim3((unsigned)M), 256, 0, stream>>>(x2, ln2_w, ln2_b, xn2, TDIM);

    // FC1 + bias + exact GELU -> h (bf16)
    gemm_wmma_kernel<0,0,1,1,0,1><<<dim3(TMLP/64, M/128, 1), 256, 0, stream>>>(
        xn2, TDIM, 0, 0,  fc1Wb, TMLP, 0, 0, nullptr,
        h16, TMLP, 0, 0,  fc1_b, 0,  nullptr, 0, 0,
        (int)M, TMLP, TDIM, 1, 1.0f);

    // FC2 (shared path) + bias + residual(x2) -> d_out cols [0,768)
    gemm_wmma_kernel<0,0,1,0,1,0><<<dim3((TDIM-TPART)/64, M/128, 1), 256, 0, stream>>>(
        h16, TMLP, 0, 0,  fc2Wb, TDIM-TPART, 0, 0, nullptr,
        outFinal, TDIM, 0, 0,  fc2_b, 0,  x2, TDIM, 0,
        (int)M, TDIM-TPART, TMLP, 1, 1.0f);

    // Expert path: per-sample gathered weights; -> d_out cols [768,1024)
    gemm_wmma_kernel<0,0,1,0,1,0><<<dim3(TPART/64, TN/128, TB), 256, 0, stream>>>(
        h16, TMLP, (long)TN*TMLP, 0,
        expWb, TPART, (long)TMLP*TPART, 0, idx,
        outFinal + (TDIM-TPART), TDIM, (long)TN*TDIM, 0,
        exp_b, TPART,
        x2 + (TDIM-TPART), TDIM, (long)TN*TDIM,
        TN, TPART, TMLP, 1, 1.0f);
}
